// Refiner_30142080483999
// MI455X (gfx1250) — matrix operand order repaired and checked by reading, not compiled, not run on hardware
//
#include <hip/hip_runtime.h>
#include <hip/hip_bf16.h>
#include <stdint.h>

// ---------------------------------------------------------------------------
// Types for CDNA5 WMMA (wave32): v_wmma_f32_16x16x32_bf16
// ---------------------------------------------------------------------------
typedef __attribute__((ext_vector_type(16))) __bf16 v16bf;
typedef __attribute__((ext_vector_type(8)))  float  v8f;
typedef __attribute__((ext_vector_type(4)))  unsigned int u32x4;
typedef __attribute__((ext_vector_type(8)))  unsigned int u32x8;

union Frag16 { v16bf v; uint4 q[2]; unsigned short s[16]; };

__device__ __forceinline__ unsigned short f2bf(float f) {
  unsigned int x = __float_as_uint(f);
  x += 0x7FFFu + ((x >> 16) & 1u);            // round-to-nearest-even
  return (unsigned short)(x >> 16);
}

// Fragment load: uniform 64-bit base (SGPR pair) + 32-bit per-lane BYTE
// offset -> GVS addressing (global_load_b128 v, voff, s[base]); the second
// 16B half folds into the instruction immediate offset.
//   A frag (16x32): gapBytes = 32.   B frag (32x16, from [N,K]): gapBytes = 32.
__device__ __forceinline__ v16bf load_frag2b(const char* __restrict__ base,
                                             unsigned boff, int gapBytes) {
  Frag16 f;
  f.q[0] = *(const uint4*)(base + boff);
  f.q[1] = *(const uint4*)(base + boff + gapBytes);
  return f.v;
}

__device__ __forceinline__ v8f wmma_bf16(v16bf a, v16bf b, v8f c) {
  return __builtin_amdgcn_wmma_f32_16x16x32_bf16(false, a, false, b, (short)0, c,
                                                 false, false);
}

// ---------------------------------------------------------------------------
// Problem constants
// ---------------------------------------------------------------------------
#define BB   32
#define TT   96
#define NN   512
#define FF   8
#define HID  512
#define MROWS (BB * NN)        // 16384
#define KIN  (TT * FF)         // 768
#define K2   (2 * HID * 9)     // 9216 combined KAN K
#define IFC  32                // in-features per KAN chunk
#define KCH  (IFC * 9)         // 288 K per chunk
#define NCH  ((2 * HID) / IFC) // 32 chunks

// ---------------------------------------------------------------------------
// Kernel 1: pack/transpose [B,T,N,F] f32 -> [B*N, T*F] bf16
// ---------------------------------------------------------------------------
__global__ void pack_kernel(const float* __restrict__ X, unsigned short* __restrict__ Xp) {
  long idx = (long)blockIdx.x * blockDim.x + threadIdx.x;   // over M*T
  if (idx >= (long)MROWS * TT) return;
  int  t = (int)(idx % TT);
  long m = idx / TT;
  int  b = (int)(m >> 9), node = (int)(m & 511);
  const float4* src = (const float4*)(X + (((long)b * TT + t) * NN + node) * FF);
  float4 a0 = src[0], a1 = src[1];
  uint4 o;
  o.x = (unsigned)f2bf(a0.x) | ((unsigned)f2bf(a0.y) << 16);
  o.y = (unsigned)f2bf(a0.z) | ((unsigned)f2bf(a0.w) << 16);
  o.z = (unsigned)f2bf(a1.x) | ((unsigned)f2bf(a1.y) << 16);
  o.w = (unsigned)f2bf(a1.z) | ((unsigned)f2bf(a1.w) << 16);
  *(uint4*)(Xp + m * KIN + t * FF) = o;
}

// ---------------------------------------------------------------------------
// Kernel 2: generic f32 -> bf16 convert (weights)
// ---------------------------------------------------------------------------
__global__ void cvt_bf16_kernel(const float* __restrict__ in, unsigned short* __restrict__ out,
                                long n) {
  long i = (long)blockIdx.x * blockDim.x + threadIdx.x;
  if (i < n) out[i] = f2bf(in[i]);
}

// ---------------------------------------------------------------------------
// Kernel 3: H[:, col0:col0+512] = tanh(A @ W^T + bias)
//   A: [M, K=768] bf16, W: [512, 768] bf16 row-major, H: [M, 1024] f32
//   block = 256 thr (8 waves = 2M x 4N), block tile 64x256,
//   wave tile 32x64 (8 WMMA per K-step, 6 frag loads).
//   2-stage ping-pong pipeline, no register rotation copies; all hot-loop
//   addresses are 32-bit byte offsets (SADDR+VOFFSET form).
// ---------------------------------------------------------------------------
__global__ void gemm_tanh_kernel(const unsigned short* __restrict__ A,
                                 const unsigned short* __restrict__ W,
                                 const float* __restrict__ bias,
                                 float* __restrict__ H,
                                 int out_ld, int out_col0) {
  const int K = KIN;                      // 768, compile-time
  const int lane = threadIdx.x & 31;
  const int wave = threadIdx.x >> 5;
  const int wm = wave & 1, wn = wave >> 1;
  const int m0 = blockIdx.x * 64 + wm * 32;
  const int n0 = blockIdx.y * 256 + wn * 64;
  const int r  = lane & 15, hh = lane >> 4;

  const char* A8 = (const char*)A;
  const char* W8 = (const char*)W;

  // per-lane 32-bit BYTE offsets (constant over the K loop)
  unsigned aoff[2], boff[4];
#pragma unroll
  for (int i = 0; i < 2; ++i) aoff[i] = ((unsigned)(m0 + i * 16 + r) * K + hh * 8) * 2u;
#pragma unroll
  for (int j = 0; j < 4; ++j) boff[j] = ((unsigned)(n0 + j * 16 + r) * K + hh * 8) * 2u;

  v8f acc[2][4];
#pragma unroll
  for (int i = 0; i < 2; ++i)
#pragma unroll
    for (int j = 0; j < 4; ++j)
#pragma unroll
      for (int q = 0; q < 8; ++q) acc[i][j][q] = 0.0f;

  v16bf a0[2], b0[4], a1[2], b1[4];       // ping-pong fragment stages

#define LOAD_STAGE(AS, BS, kb)                                              \
  {                                                                         \
    _Pragma("unroll")                                                       \
    for (int i = 0; i < 2; ++i) AS[i] = load_frag2b(A8, aoff[i] + (kb), 32);\
    _Pragma("unroll")                                                       \
    for (int j = 0; j < 4; ++j) BS[j] = load_frag2b(W8, boff[j] + (kb), 32);\
  }
#define WMMA_STAGE(AS, BS)                                                  \
  {                                                                         \
    _Pragma("unroll")                                                       \
    for (int i = 0; i < 2; ++i)                                             \
      _Pragma("unroll")                                                     \
      for (int j = 0; j < 4; ++j) acc[i][j] = wmma_bf16(AS[i], BS[j], acc[i][j]); \
  }

  LOAD_STAGE(a0, b0, 0u);
  // 24 K-steps total (K/32); 11 full pairs + tail pair
  for (unsigned kb = 64; kb <= (unsigned)(K - 64) * 2; kb += 128) {
    LOAD_STAGE(a1, b1, kb);          // bytes: k-step at kb/2 elements
    WMMA_STAGE(a0, b0);
    LOAD_STAGE(a0, b0, kb + 64);
    WMMA_STAGE(a1, b1);
  }
  LOAD_STAGE(a1, b1, (unsigned)(K - 32) * 2);
  WMMA_STAGE(a0, b0);
  WMMA_STAGE(a1, b1);
#undef LOAD_STAGE
#undef WMMA_STAGE

  const int r16 = lane & 15;
#pragma unroll
  for (int i = 0; i < 2; ++i)
#pragma unroll
    for (int j = 0; j < 4; ++j)
#pragma unroll
      for (int q = 0; q < 8; ++q) {
        int row = m0 + i * 16 + q + hh * 8;
        int col = n0 + j * 16 + r16;
        H[(long)row * out_ld + out_col0 + col] = tanhf(acc[i][j][q] + bias[col]);
      }
}

// ---------------------------------------------------------------------------
// Kernel 4: build combined KAN weight Bk [64, 9216] bf16:
//   Bk[o, i*9+0]   = base_w[o,i]
//   Bk[o, i*9+1+j] = spline_w[o,i,j] * spline_scaler[o,i]
// ---------------------------------------------------------------------------
__global__ void kan_prep_w_kernel(const float* __restrict__ base_w,
                                  const float* __restrict__ spline_w,
                                  const float* __restrict__ scaler,
                                  unsigned short* __restrict__ Bk) {
  int idx = blockIdx.x * blockDim.x + threadIdx.x;   // 64 * 1024
  if (idx >= 64 * 1024) return;
  int o = idx >> 10, i = idx & 1023;
  unsigned short* dst = Bk + (long)o * K2 + i * 9;
  dst[0] = f2bf(base_w[idx]);
  float sc = scaler[idx];
  const float* sw = spline_w + (long)idx * 8;
#pragma unroll
  for (int j = 0; j < 8; ++j) dst[1 + j] = f2bf(sw[j] * sc);
}

// ---------------------------------------------------------------------------
// TDM: issue a 2D tensor_load_to_lds for a [64 rows x 288 elem] bf16 tile of
// Bk (row stride 9216 elems) into LDS at ldsAddr. Wave-uniform; tracked by
// TENSORcnt. D# per CDNA5 ISA ch.8.
// ---------------------------------------------------------------------------
__device__ __forceinline__ void tdm_load_bk_chunk(const unsigned short* gptr,
                                                  unsigned int ldsAddr) {
  unsigned long long ga = (unsigned long long)(uintptr_t)gptr;
  u32x4 g0;
  g0[0] = 1u;                                            // count=1, user mode
  g0[1] = ldsAddr;                                       // lds_addr (bytes)
  g0[2] = (unsigned int)(ga & 0xFFFFFFFFull);            // global_addr[31:0]
  g0[3] = (unsigned int)((ga >> 32) & 0x01FFFFFFull)     // global_addr[56:32]
        | 0x80000000u;                                   // type=2 ("image")
  u32x8 g1;
  g1[0] = 1u << 16;          // data_size = 1 (2 bytes); no multicast/pad/iter
  g1[1] = (unsigned)(K2 & 0xFFFF) << 16;   // tensor_dim0[15:0] @ bits 63:48
  g1[2] = (unsigned)(K2 >> 16)             // tensor_dim0[31:16]
        | (64u << 16);                     // tensor_dim1[15:0] @ bits 95:80
  g1[3] = 0u                               // tensor_dim1[31:16]
        | ((unsigned)KCH << 16);           // tile_dim0 @ bits 127:112
  g1[4] = 64u;                             // tile_dim1; tile_dim2 = 0
  g1[5] = (unsigned)K2;                    // tensor_dim0_stride[31:0]
  g1[6] = 0u;                              // stride hi / dim1_stride lo
  g1[7] = 0u;
  asm volatile("tensor_load_to_lds %0, %1" :: "s"(g0), "s"(g1) : "memory");
}

// ---------------------------------------------------------------------------
// Kernel 5: fused KAN GEMM. kan[m,o] = sum_i silu(h)*base_w + sum_ij B_ij*sw
// A tile (silu + 8 de Boor bases per in-feature) generated in LDS per chunk.
// B chunk (36KB) DMA'd into LDS by the Tensor Data Mover, double-buffered so
// the transfer overlaps the generate phase + WMMAs of the previous chunk.
// block = 256 thr (8 waves), out tile 32x64, 32 chunks x 9 WMMA steps.
// dynamic LDS: 32*288*2 (A) + 2 * 64*288*2 (B ping/pong) = 92160 bytes.
// ---------------------------------------------------------------------------
__global__ void kan_kernel(const float* __restrict__ Hbuf,          // [M, 1024]
                           const unsigned short* __restrict__ Bk,   // [64, 9216]
                           unsigned short* __restrict__ kanbf) {    // [M, 64] bf16
  extern __shared__ unsigned short smem[];
  unsigned short* ldsA = smem;                     // [32][KCH]
  unsigned short* ldsB = smem + 32 * KCH;          // [2][64][KCH]

  const int lane = threadIdx.x & 31;
  const int wave = threadIdx.x >> 5;
  const int wm = wave & 1, wn = wave >> 1;
  const int mblk = blockIdx.x * 32;

  v8f acc;
#pragma unroll
  for (int q = 0; q < 8; ++q) acc[q] = 0.0f;

  // kick off DMA of B chunk 0 into buffer 0
  if (wave == 0)
    tdm_load_bk_chunk(Bk, (unsigned int)(uintptr_t)(ldsB));

  for (int ch = 0; ch < NCH; ++ch) {
    const int buf = ch & 1;
    // ---- generate phase: 32 rows x 32 in-features -> 9 bf16 each ----
    // (VALU work overlaps the in-flight TDM transfer of B[ch])
    {
      int rowl = threadIdx.x >> 3;         // 0..31
      int ig   = (threadIdx.x & 7) * 4;    // 0..28 step 4
      const float* hp = Hbuf + (long)(mblk + rowl) * (2 * HID) + ch * IFC + ig;
#pragma unroll
      for (int u = 0; u < 4; ++u) {
        float x = hp[u];
        unsigned short* dst = &ldsA[rowl * KCH + (ig + u) * 9];
        dst[0] = f2bf(x / (1.0f + __expf(-x)));           // silu
        float bas[11];
#pragma unroll
        for (int i = 0; i < 11; ++i) {                    // order-0 indicators
          float t0 = -2.2f + 0.4f * i;
          bas[i] = (x >= t0 && x < t0 + 0.4f) ? 1.0f : 0.0f;
        }
#pragma unroll
        for (int k = 1; k <= 3; ++k) {                    // de Boor-Cox
          float inv = 1.0f / (0.4f * k);
#pragma unroll
          for (int i = 0; i < 10; ++i) {
            if (i < 11 - k) {
              float ti = -2.2f + 0.4f * i;
              float l  = (x - ti) * inv;
              float rr = ((ti + 0.4f * (k + 1)) - x) * inv;
              bas[i] = l * bas[i] + rr * bas[i + 1];
            }
          }
        }
#pragma unroll
        for (int j = 0; j < 8; ++j) dst[1 + j] = f2bf(bas[j]);
      }
    }
    if (wave == 0) __builtin_amdgcn_s_wait_tensorcnt(0);  // B[ch] landed
    __syncthreads();                                      // A + B[ch] visible

    // queue DMA for next chunk into the other buffer; overlaps WMMA phase
    if (wave == 0 && ch + 1 < NCH)
      tdm_load_bk_chunk(Bk + (long)(ch + 1) * KCH,
                        (unsigned int)(uintptr_t)(ldsB + (1 - buf) * 64 * KCH));

    // ---- WMMA phase: 9 k-steps of 32, A and B both from LDS ----
    const int r = lane & 15, hh = lane >> 4;
    const unsigned short* bbase = ldsB + buf * 64 * KCH;
#pragma unroll
    for (int ks = 0; ks < 9; ++ks) {
      Frag16 fa, fb;
      const unsigned short* ap = &ldsA[(wm * 16 + r) * KCH + ks * 32 + hh * 8];
      fa.q[0] = *(const uint4*)(ap);
      fa.q[1] = *(const uint4*)(ap + 16);
      const unsigned short* bp = &bbase[(wn * 16 + r) * KCH + ks * 32 + hh * 8];
      fb.q[0] = *(const uint4*)(bp);
      fb.q[1] = *(const uint4*)(bp + 16);
      acc = wmma_bf16(fa.v, fb.v, acc);
    }
    __syncthreads();   // protect ldsA before next generate phase
  }

  const int r16 = lane & 15, hh = lane >> 4;
#pragma unroll
  for (int q = 0; q < 8; ++q) {
    int row = mblk + wm * 16 + q + hh * 8;
    kanbf[(long)row * 64 + wn * 16 + r16] = f2bf(acc[q]);
  }
}

// ---------------------------------------------------------------------------
// Kernel 6: out = kan @ Wf^T + bf, scattered to [B, T_OUT, N, F]
//   kan: [M, 64] bf16; Wf: [768, 64] bf16 (already [N_out, K] row-major)
// ---------------------------------------------------------------------------
__global__ void final_kernel(const unsigned short* __restrict__ A,    // [M,64]
                             const unsigned short* __restrict__ Wfb,  // [768,64]
                             const float* __restrict__ bfv,           // [768]
                             float* __restrict__ out) {
  const int lane = threadIdx.x & 31;
  const int wave = threadIdx.x >> 5;
  const int wm = wave & 1, wn = wave >> 1;
  const int m0 = blockIdx.x * 32 + wm * 16;
  const int n0 = blockIdx.y * 256 + wn * 64;
  const int r  = lane & 15, hh = lane >> 4;

  const char* A8 = (const char*)A;
  const char* W8 = (const char*)Wfb;
  unsigned aoff = ((unsigned)(m0 + r) * 64 + hh * 8) * 2u;
  unsigned boff[4];
#pragma unroll
  for (int j = 0; j < 4; ++j) boff[j] = ((unsigned)(n0 + j * 16 + r) * 64 + hh * 8) * 2u;

  v8f acc[4];
#pragma unroll
  for (int j = 0; j < 4; ++j)
#pragma unroll
    for (int q = 0; q < 8; ++q) acc[j][q] = 0.0f;

#pragma unroll
  for (unsigned kb = 0; kb < 128; kb += 64) {    // two K-steps of 32 (bytes)
    v16bf a = load_frag2b(A8, aoff + kb, 32);
#pragma unroll
    for (int j = 0; j < 4; ++j) {
      v16bf b = load_frag2b(W8, boff[j] + kb, 32);
      acc[j] = wmma_bf16(a, b, acc[j]);
    }
  }

  const int r16 = lane & 15;
#pragma unroll
  for (int j = 0; j < 4; ++j)
#pragma unroll
    for (int q = 0; q < 8; ++q) {
      int row  = m0 + q + hh * 8;
      int col  = n0 + j * 16 + r16;     // 0..767
      int b    = row >> 9, node = row & 511;
      int t    = col >> 3, f    = col & 7;
      out[(((long)b * TT + t) * NN + node) * FF + f] = acc[j][q] + bfv[col];
    }
}

// ---------------------------------------------------------------------------
// Host launch
// ---------------------------------------------------------------------------
extern "C" void kernel_launch(void* const* d_in, const int* in_sizes, int n_in,
                              void* d_out, int out_size, void* d_ws, size_t ws_size,
                              hipStream_t stream) {
  const float* X       = (const float*)d_in[0];
  const float* Y       = (const float*)d_in[1];
  const float* Wx      = (const float*)d_in[2];
  const float* bx      = (const float*)d_in[3];
  const float* Wy      = (const float*)d_in[4];
  const float* by      = (const float*)d_in[5];
  const float* base_w  = (const float*)d_in[6];
  const float* splinew = (const float*)d_in[7];
  const float* scaler  = (const float*)d_in[8];
  const float* Wf      = (const float*)d_in[9];
  const float* bfv     = (const float*)d_in[10];
  float* out = (float*)d_out;

  char* ws = (char*)d_ws;
  size_t off = 0;
  auto carve = [&](size_t bytes) -> char* {
    char* p = ws + off;
    off = (off + bytes + 255) & ~(size_t)255;
    return p;
  };
  unsigned short* Xp    = (unsigned short*)carve((size_t)MROWS * KIN * 2);
  unsigned short* Yp    = (unsigned short*)carve((size_t)MROWS * KIN * 2);
  unsigned short* Wxb   = (unsigned short*)carve((size_t)HID * KIN * 2);
  unsigned short* Wyb   = (unsigned short*)carve((size_t)HID * KIN * 2);
  unsigned short* Wfb   = (unsigned short*)carve((size_t)KIN * 64 * 2);
  float*          H     = (float*)carve((size_t)MROWS * 2 * HID * 4);
  unsigned short* Bk    = (unsigned short*)carve((size_t)64 * K2 * 2);
  unsigned short* kanbf = (unsigned short*)carve((size_t)MROWS * 64 * 2);

  const int T = 256;
  long packN = (long)MROWS * TT;
  pack_kernel<<<dim3((unsigned)((packN + T - 1) / T)), T, 0, stream>>>(X, Xp);
  pack_kernel<<<dim3((unsigned)((packN + T - 1) / T)), T, 0, stream>>>(Y, Yp);

  long wxn = (long)HID * KIN;
  cvt_bf16_kernel<<<dim3((unsigned)((wxn + T - 1) / T)), T, 0, stream>>>(Wx, Wxb, wxn);
  cvt_bf16_kernel<<<dim3((unsigned)((wxn + T - 1) / T)), T, 0, stream>>>(Wy, Wyb, wxn);
  long wfn = (long)KIN * 64;
  cvt_bf16_kernel<<<dim3((unsigned)((wfn + T - 1) / T)), T, 0, stream>>>(Wf, Wfb, wfn);

  // Hx -> H[:, 0:512], Hy -> H[:, 512:1024]
  gemm_tanh_kernel<<<dim3(MROWS / 64, HID / 256), T, 0, stream>>>(Xp, Wxb, bx, H, 2 * HID, 0);
  gemm_tanh_kernel<<<dim3(MROWS / 64, HID / 256), T, 0, stream>>>(Yp, Wyb, by, H, 2 * HID, HID);

  kan_prep_w_kernel<<<dim3((64 * 1024) / T), T, 0, stream>>>(base_w, splinew, scaler, Bk);

  const size_t kanLds = (size_t)(32 * KCH + 2 * 64 * KCH) * sizeof(unsigned short);
  kan_kernel<<<dim3(MROWS / 32), T, kanLds, stream>>>(H, Bk, kanbf);

  final_kernel<<<dim3(MROWS / 32, (TT * FF) / 256), T, 0, stream>>>(kanbf, Wfb, bfv, out);

  (void)in_sizes; (void)n_in; (void)out_size; (void)ws_size;
}
